// SageModel_59682865545779
// MI455X (gfx1250) — compile-verified, matrix operand order checked by reading, not run on hardware
//
#include <hip/hip_runtime.h>
#include <hip/hip_bf16.h>
#include <math.h>

typedef __attribute__((ext_vector_type(16))) _Float16 v16h;
typedef __attribute__((ext_vector_type(8)))  float    v8f;

// Branch-free tanh: 1 - 2/(exp(2v)+1). Uses v_exp_f32 (TRANS) + v_rcp_f32.
// Saturates correctly to +/-1 for large |v|; ~1e-7 abs error, far below the
// f16-input WMMA noise floor. Avoids libm tanhf's divergent range-split.
__device__ __forceinline__ float fast_tanh(float v) {
    float e = __expf(2.0f * v);
    return 1.0f - 2.0f * __builtin_amdgcn_rcpf(e + 1.0f);
}

// ---------------------------------------------------------------- utilities

__global__ void zero_f32_kernel(float* __restrict__ p, int n) {
    int i = blockIdx.x * blockDim.x + threadIdx.x;
    int stride = gridDim.x * blockDim.x;
    for (; i < n; i += stride) p[i] = 0.0f;
}

__global__ void degree_kernel(const int* __restrict__ dst,
                              float* __restrict__ deg, int E) {
    int e = blockIdx.x * blockDim.x + threadIdx.x;
    if (e < E) atomicAdd(&deg[dst[e]], 1.0f);
}

__global__ void invdeg_kernel(const float* __restrict__ deg,
                              float* __restrict__ invdeg, int n) {
    int i = blockIdx.x * blockDim.x + threadIdx.x;
    if (i < n) invdeg[i] = 1.0f / fmaxf(deg[i], 1.0f);
}

// Pack stacked weights [Wl; Wr] (f32 row-major [Fin,Fout] each) into f16
// WMMA B-fragment order:
//   out[(((tn*nC + c)*32) + lane)*16 + j] = Wcat[k][col]
//     k   = c*32 + ((j<8)? j : j+8) + (lane>>4)*8
//     col = tn*16 + (lane&15)
// so each lane's 16-half fragment for (tn, chunk c) is one contiguous 32B load.
__global__ void pack_w16_kernel(const float* __restrict__ Wl,
                                const float* __restrict__ Wr,
                                _Float16* __restrict__ out,
                                int Fin, int K, int Fout) {
    int i = blockIdx.x * blockDim.x + threadIdx.x;
    int total = K * Fout;
    if (i >= total) return;
    const int nC   = K >> 5;
    const int j    = i & 15;
    const int lane = (i >> 4) & 31;
    const int rest = i >> 9;
    const int c    = rest % nC;
    const int tn   = rest / nC;
    const int k    = c * 32 + ((j < 8) ? j : (j + 8)) + (lane >> 4) * 8;
    const int col  = tn * 16 + (lane & 15);
    float v = (k < Fin) ? Wl[k * Fout + col] : Wr[(k - Fin) * Fout + col];
    out[i] = (_Float16)v;
}

// Edge-parallel scatter-add with inv_deg folded in:
//   agg[d][f] += xcat[s][f] * invdeg[d]
// (valid since invdeg[d] is constant per destination row)
// xcat = [xa (Fa) | xb (F-Fa)], F = 1<<logF. Fa multiple of 32 -> branch is
// wave-uniform (a wave covers 32 consecutive f values).
__global__ void scatter_add_kernel(const int* __restrict__ src,
                                   const int* __restrict__ dst,
                                   const float* __restrict__ xa, int Fa,
                                   const float* __restrict__ xb, int logF,
                                   const float* __restrict__ invdeg,
                                   float* __restrict__ agg, int total) {
    int i = blockIdx.x * blockDim.x + threadIdx.x;
    if (i >= total) return;
    const int F = 1 << logF;
    const int e = i >> logF;
    const int f = i & (F - 1);
    const int s = src[e];
    const int d = dst[e];
    float v = (f < Fa) ? xa[s * Fa + f] : xb[s * (F - Fa) + (f - Fa)];
    atomicAdd(&agg[d * F + f], v * invdeg[d]);
}

// ---------------------------------------------------------------- WMMA GEMM
//
// out[m][n] = act( sum_k A[m][k] * Wcat[k][n] + bias[n] )
//   A = [agg(aggF, already inv_deg-scaled) | xa(Fa) | xb(Fb)]   (virtual concat)
// K = aggF+Fa+Fb (all multiples of 32). Fout multiple of 32. tilesM = N/16 exact.
// One wave computes a 16x32 C slab: one A fragment feeds two
// v_wmma_f32_16x16x32_f16 per K chunk. A gather = 4x b128 loads (uniform
// source per chunk); B fragment = one contiguous 32B pre-packed load.

__global__ __launch_bounds__(256)
void sage_gemm_wmma_kernel(const float* __restrict__ agg, int aggF,
                           const float* __restrict__ xa, int Fa,
                           const float* __restrict__ xb, int Fb,
                           const v16h* __restrict__ Wp,   // packed fragments
                           const float* __restrict__ bias,
                           float* __restrict__ out,
                           int tilesM, int Fout, int doTanh) {
    const int K      = aggF + Fa + Fb;
    const int nC     = K >> 5;
    const int pairsN = Fout >> 5;
    const int total  = tilesM * pairsN;
    const int wid    = blockIdx.x * (blockDim.x >> 5) + (threadIdx.x >> 5);
    if (wid >= total) return;                 // uniform per wave

    const int lane = threadIdx.x & 31;
    const int hi   = lane >> 4;
    const int lid  = lane & 15;
    const int tm   = wid / pairsN;
    const int pr   = wid - tm * pairsN;
    const int row  = tm * 16 + lid;           // A row this lane gathers
    const int col  = pr * 32 + lid;           // first output column

    const v16h* __restrict__ Wp0 = Wp + ((size_t)(2 * pr)     * nC) * 32 + lane;
    const v16h* __restrict__ Wp1 = Wp + ((size_t)(2 * pr + 1) * nC) * 32 + lane;

    v8f acc0 = {}, acc1 = {};
    for (int c = 0; c < nC; ++c) {
        const int k0 = c * 32;
        // chunk source: uniform per wave (aggF/Fa multiples of 32)
        const float* sp; int sF, kb;
        if (k0 < aggF)           { sp = agg; sF = aggF; kb = k0; }
        else if (k0 < aggF + Fa) { sp = xa;  sF = Fa;   kb = k0 - aggF; }
        else                     { sp = xb;  sF = Fb;   kb = k0 - aggF - Fa; }

        const float* rp = sp + (size_t)row * sF + kb + hi * 8;
        const float4 a0 = *(const float4*)(rp);        // K = kb+hi*8   +0..3
        const float4 a1 = *(const float4*)(rp + 4);    //               +4..7
        const float4 a2 = *(const float4*)(rp + 16);   // K = kb+16+hi*8+0..3
        const float4 a3 = *(const float4*)(rp + 20);   //               +4..7

        v16h aF;
        aF[0]  = (_Float16)a0.x; aF[1]  = (_Float16)a0.y;
        aF[2]  = (_Float16)a0.z; aF[3]  = (_Float16)a0.w;
        aF[4]  = (_Float16)a1.x; aF[5]  = (_Float16)a1.y;
        aF[6]  = (_Float16)a1.z; aF[7]  = (_Float16)a1.w;
        aF[8]  = (_Float16)a2.x; aF[9]  = (_Float16)a2.y;
        aF[10] = (_Float16)a2.z; aF[11] = (_Float16)a2.w;
        aF[12] = (_Float16)a3.x; aF[13] = (_Float16)a3.y;
        aF[14] = (_Float16)a3.z; aF[15] = (_Float16)a3.w;

        const v16h bF0 = Wp0[c * 32];
        const v16h bF1 = Wp1[c * 32];

        acc0 = __builtin_amdgcn_wmma_f32_16x16x32_f16(
                   false, aF, false, bF0, (short)0, acc0, false, false);
        acc1 = __builtin_amdgcn_wmma_f32_16x16x32_f16(
                   false, aF, false, bF1, (short)0, acc1, false, false);
    }

    const float b0 = bias[col];
    const float b1 = bias[col + 16];
#pragma unroll
    for (int r = 0; r < 8; ++r) {
        const int m = tm * 16 + r + 8 * hi;
        float v0 = acc0[r] + b0;
        float v1 = acc1[r] + b1;
        if (doTanh) { v0 = fast_tanh(v0); v1 = fast_tanh(v1); }
        out[(size_t)m * Fout + col]      = v0;
        out[(size_t)m * Fout + col + 16] = v1;
    }
}

// Final projection: out = [h2(32) | skip2(32)] @ dec_W(64x3) + dec_b
__global__ void dec_kernel(const float* __restrict__ h2,
                           const float* __restrict__ skip2,
                           const float* __restrict__ W,   // [64][3] f32
                           const float* __restrict__ b,   // [3]
                           float* __restrict__ out, int n) {
    int i = blockIdx.x * blockDim.x + threadIdx.x;
    if (i >= n) return;
    float a0 = b[0], a1 = b[1], a2 = b[2];
#pragma unroll 4
    for (int k = 0; k < 32; ++k) {
        float v = h2[i * 32 + k];
        a0 += v * W[k * 3 + 0];
        a1 += v * W[k * 3 + 1];
        a2 += v * W[k * 3 + 2];
    }
#pragma unroll 4
    for (int k = 0; k < 32; ++k) {
        float v = skip2[i * 32 + k];
        a0 += v * W[(32 + k) * 3 + 0];
        a1 += v * W[(32 + k) * 3 + 1];
        a2 += v * W[(32 + k) * 3 + 2];
    }
    out[i * 3 + 0] = a0;
    out[i * 3 + 1] = a1;
    out[i * 3 + 2] = a2;
}

// ---------------------------------------------------------------- launcher

extern "C" void kernel_launch(void* const* d_in, const int* in_sizes, int n_in,
                              void* d_out, int out_size, void* d_ws, size_t ws_size,
                              hipStream_t stream) {
    const float* x     = (const float*)d_in[0];
    const int*   edge  = (const int*)d_in[1];
    const float* enc_W = (const float*)d_in[2];
    const float* enc_b = (const float*)d_in[3];
    const float* c1_Wl = (const float*)d_in[4];
    const float* c1_bl = (const float*)d_in[5];
    const float* c1_Wr = (const float*)d_in[6];
    const float* c2_Wl = (const float*)d_in[7];
    const float* c2_bl = (const float*)d_in[8];
    const float* c2_Wr = (const float*)d_in[9];
    const float* bn_Wl = (const float*)d_in[10];
    const float* bn_bl = (const float*)d_in[11];
    const float* bn_Wr = (const float*)d_in[12];
    const float* d1_Wl = (const float*)d_in[13];
    const float* d1_bl = (const float*)d_in[14];
    const float* d1_Wr = (const float*)d_in[15];
    const float* d2_Wl = (const float*)d_in[16];
    const float* d2_bl = (const float*)d_in[17];
    const float* d2_Wr = (const float*)d_in[18];
    const float* s1_W  = (const float*)d_in[19];
    const float* s1_b  = (const float*)d_in[20];
    const float* s2_W  = (const float*)d_in[21];
    const float* s2_b  = (const float*)d_in[22];
    const float* dec_W = (const float*)d_in[23];
    const float* dec_b = (const float*)d_in[24];
    float* out = (float*)d_out;

    const int NN = in_sizes[0] / 64;      // 100000 (divisible by 16)
    const int EE = in_sizes[1] / 2;       // 1600000
    const int tilesM = NN / 16;
    const int* src = edge;
    const int* dst = edge + EE;

    // ---- workspace carve-up (all written before read) ----
    char* ws = (char*)d_ws;
    size_t off = 0;
    auto allocF = [&](size_t nf) { float* p = (float*)(ws + off); off += nf * sizeof(float); return p; };
    float* deg    = allocF((size_t)NN);
    float* invdeg = allocF((size_t)NN);
    float* x0     = allocF((size_t)NN * 32);
    float* x1     = allocF((size_t)NN * 64);
    float* x2     = allocF((size_t)NN * 128);
    float* x3     = allocF((size_t)NN * 128);
    float* skip1  = allocF((size_t)NN * 64);
    float* skip2  = allocF((size_t)NN * 32);
    float* h1     = allocF((size_t)NN * 64);
    float* h2     = allocF((size_t)NN * 32);
    float* agg    = allocF((size_t)NN * 256);
    auto allocH = [&](size_t nh) { _Float16* p = (_Float16*)(ws + off); off += nh * sizeof(_Float16); return p; };
    _Float16* w_enc = allocH(64 * 32);
    _Float16* w_c1  = allocH(64 * 64);
    _Float16* w_c2  = allocH(128 * 128);
    _Float16* w_bn  = allocH(256 * 128);
    _Float16* w_d1  = allocH(512 * 64);
    _Float16* w_d2  = allocH(256 * 32);
    _Float16* w_s1  = allocH(64 * 64);
    _Float16* w_s2  = allocH(32 * 32);
    (void)ws_size; (void)n_in; (void)out_size;

    const int T = 256;
    auto blk = [&](int n) { return (n + T - 1) / T; };

    // ---- degrees ----
    zero_f32_kernel<<<blk(NN), T, 0, stream>>>(deg, NN);
    degree_kernel<<<blk(EE), T, 0, stream>>>(dst, deg, EE);
    invdeg_kernel<<<blk(NN), T, 0, stream>>>(deg, invdeg, NN);

    // ---- f16 weight prep (stack [Wl;Wr] + pack to fragment order) ----
    auto packw = [&](const float* Wl, const float* Wr, _Float16* o, int Fin, int K, int Fout) {
        pack_w16_kernel<<<blk(K * Fout), T, 0, stream>>>(Wl, Wr, o, Fin, K, Fout);
    };
    packw(enc_W, nullptr, w_enc, 64, 64, 32);
    packw(c1_Wl, c1_Wr,   w_c1,  32, 64, 64);
    packw(c2_Wl, c2_Wr,   w_c2,  64, 128, 128);
    packw(bn_Wl, bn_Wr,   w_bn,  128, 256, 128);
    packw(d1_Wl, d1_Wr,   w_d1,  256, 512, 64);
    packw(d2_Wl, d2_Wr,   w_d2,  128, 256, 32);
    packw(s1_W,  nullptr, w_s1,  64, 64, 64);
    packw(s2_W,  nullptr, w_s2,  32, 32, 32);

    // ---- helpers ----
    auto gemm = [&](const float* ag, int aggF, const float* xa, int Fa,
                    const float* xb, int Fb, const _Float16* W, const float* b,
                    float* o, int Fout, int act) {
        int tiles = tilesM * (Fout / 32);      // 16x32 slab per wave
        int blocks = (tiles + 7) / 8;          // 8 waves / 256-thread block
        sage_gemm_wmma_kernel<<<blocks, 256, 0, stream>>>(
            ag, aggF, xa, Fa, xb, Fb, (const v16h*)W, b, o, tilesM, Fout, act);
    };
    auto aggregate = [&](const float* xa, int Fa, const float* xb, int logF) {
        int F = 1 << logF;
        zero_f32_kernel<<<blk(NN * F), T, 0, stream>>>(agg, NN * F);
        int total = EE << logF;
        scatter_add_kernel<<<blk(total), T, 0, stream>>>(src, dst, xa, Fa, xb, logF, invdeg, agg, total);
    };

    // ---- network ----
    // x0 = tanh(x @ enc_W + enc_b)                       [N,32]
    gemm(nullptr, 0, x, 64, nullptr, 0, w_enc, enc_b, x0, 32, 1);
    // skip2 = x0 @ s2_W + s2_b (no tanh)                 [N,32]
    gemm(nullptr, 0, x0, 32, nullptr, 0, w_s2, s2_b, skip2, 32, 0);

    // x1 = tanh(sage(x0))                                [N,64]
    aggregate(x0, 32, nullptr, 5);
    gemm(agg, 32, x0, 32, nullptr, 0, w_c1, c1_bl, x1, 64, 1);
    // skip1 = x1 @ s1_W + s1_b                           [N,64]
    gemm(nullptr, 0, x1, 64, nullptr, 0, w_s1, s1_b, skip1, 64, 0);

    // x2 = tanh(sage(x1))                                [N,128]
    aggregate(x1, 64, nullptr, 6);
    gemm(agg, 64, x1, 64, nullptr, 0, w_c2, c2_bl, x2, 128, 1);

    // x3 = tanh(sage(x2))                                [N,128]
    aggregate(x2, 128, nullptr, 7);
    gemm(agg, 128, x2, 128, nullptr, 0, w_bn, bn_bl, x3, 128, 1);

    // h1 = tanh(sage([x3|x2]))                           [N,64]
    aggregate(x3, 128, x2, 8);
    gemm(agg, 256, x3, 128, x2, 128, w_d1, d1_bl, h1, 64, 1);

    // h2 = tanh(sage([h1|skip1]))                        [N,32]
    aggregate(h1, 64, skip1, 7);
    gemm(agg, 128, h1, 64, skip1, 64, w_d2, d2_bl, h2, 32, 1);

    // out = [h2|skip2] @ dec_W + dec_b                   [N,3]
    dec_kernel<<<blk(NN), T, 0, stream>>>(h2, skip2, dec_W, dec_b, out, NN);
}
